// FI_ERNN_40742059770159
// MI455X (gfx1250) — matrix operand order repaired and checked by reading, not compiled, not run on hardware
//
#include <hip/hip_runtime.h>

typedef __attribute__((ext_vector_type(16))) _Float16 v16h;
typedef __attribute__((ext_vector_type(8)))  _Float16 v8h;
typedef __attribute__((ext_vector_type(8)))  float    v8f;
typedef __attribute__((ext_vector_type(4)))  float    v4f;

#define U_DIM 128
#define GAMMA 28
#define WIN 49
#define BATCH 16384
#define WAVES 4
#define ROWS_PER_WAVE 16
#define ROWS_PER_BLOCK (WAVES * ROWS_PER_WAVE)   // 64
#define NFRAG 96        // 8 n-groups * 3 gates * 4 k-fragments
#define FRAG_HALF 512   // 32 lanes * 16 halves per B fragment
// fragment id: (j*3 + g)*4 + kf  -> one j-group's 12 fragments contiguous (12 KB)
#define FID(j, g, kf) ((((j) * 3 + (g)) * 4 + (kf)) * FRAG_HALF)

__device__ __forceinline__ float fsigmoid(float v) {
    return 1.0f / (1.0f + __expf(-v));
}
__device__ __forceinline__ float ftanh(float v) {
    return 2.0f / (1.0f + __expf(-2.0f * v)) - 1.0f;
}
__device__ __forceinline__ float fsoftplus(float v) {
    return fmaxf(v, 0.0f) + __logf(1.0f + __expf(-fabsf(v)));
}

__global__ __launch_bounds__(WAVES * 32, 1)
void fi_ernn_kernel(const float* __restrict__ x,    // (B,49,1)
                    const float* __restrict__ gk,   // (1,384)
                    const float* __restrict__ grk,  // (128,384)
                    const float* __restrict__ gb,   // (2,384)
                    const float* __restrict__ ek,   // (28,128,4)
                    const float* __restrict__ eb,   // (28,4)
                    float* __restrict__ out)        // (B,28,4)
{
    extern __shared__ v16h smem_v[];  // 32B-aligned dynamic LDS
    _Float16* wlds = (_Float16*)smem_v;                                   // 96 KB swizzled W_rec
    _Float16* hall = (_Float16*)((char*)smem_v + NFRAG * FRAG_HALF * 2);  // 16 KB per-wave h (f16, row-major)
    float*    hf32 = (float*)((char*)hall + WAVES * ROWS_PER_WAVE * U_DIM * 2); // 32 KB per-wave h (f32 carry)
    float*    cst  = (float*)((char*)hf32 + WAVES * ROWS_PER_WAVE * U_DIM * 4); // 4 KB gate consts (col x 8)
    float*    xall = (float*)((char*)cst + U_DIM * 8 * 4);                      // 256 B x feed

    const int tid  = threadIdx.x;
    const int wave = tid >> 5;
    const int lane = tid & 31;
    const int n16  = lane & 15;       // N index within tile / A-row M
    const int hb   = lane >> 4;       // half-wave: 0 or 1

    _Float16* hlds = hall + wave * (ROWS_PER_WAVE * U_DIM);
    float*    hfw  = hf32 + wave * (ROWS_PER_WAVE * U_DIM);
    float*    xcur = xall + wave * ROWS_PER_WAVE;
    const _Float16* wl = wlds + lane * 16;   // per-lane B-fragment base

    // ---- One-time: swizzle W_rec (f32, 128x384 row-major) into WMMA B-fragment order, f16 ----
    for (int idx = tid; idx < NFRAG * FRAG_HALF; idx += blockDim.x) {
        int f  = idx >> 9;            // fragment id = (j*3 + g)*4 + kf
        int ln = (idx >> 4) & 31;
        int i  = idx & 15;
        int kf = f & 3;
        int jg = f >> 2;              // j*3 + g
        int g  = jg % 3;
        int j  = jg / 3;
        int k   = kf * 32 + ((i & 8) ? 16 : 0) + ((ln & 16) ? 8 : 0) + (i & 7);  // B 32x16: lanes 0-15 K 0..7/16..23, lanes 16-31 K 8..15/24..31
        int col = g * U_DIM + j * 16 + (ln & 15);
        wlds[idx] = (_Float16)grk[k * 384 + col];
    }
    for (int idx = tid; idx < WAVES * ROWS_PER_WAVE * U_DIM; idx += blockDim.x) {
        hall[idx] = (_Float16)0.0f;
        hf32[idx] = 0.0f;
    }
    // ---- One-time: packed per-column gate constants (8 floats per column, 32B) ----
    for (int c = tid; c < U_DIM; c += blockDim.x) {
        cst[c * 8 + 0] = gk[c];                                   // kz
        cst[c * 8 + 1] = gk[U_DIM + c];                           // kr
        cst[c * 8 + 2] = gk[2 * U_DIM + c];                       // kh
        cst[c * 8 + 3] = gb[c] + gb[384 + c];                     // sz = b0z + b1z
        cst[c * 8 + 4] = gb[U_DIM + c] + gb[384 + U_DIM + c];     // sr = b0r + b1r
        cst[c * 8 + 5] = gb[2 * U_DIM + c];                       // b0h
        cst[c * 8 + 6] = gb[384 + 2 * U_DIM + c];                 // b1h
        cst[c * 8 + 7] = 0.0f;
    }
    __syncthreads();

    const float* cbase = cst + n16 * 8;  // per-lane constant base (column j*16 + n16)
    const int rowbase = blockIdx.x * ROWS_PER_BLOCK + wave * ROWS_PER_WAVE;

    // ---- One GRU step ----
    auto gru_step = [&]() {
        float xv[8];
        #pragma unroll
        for (int e = 0; e < 8; ++e) xv[e] = xcur[e + hb * 8];

        // A fragments: ISA 16-bit A 16x32 layout.
        // lane<16: M=lane, i0..7 -> K 0..7, i8..15 -> K 16..23; lane>=16: K 8..15 / 24..31.
        v16h a[4];
        const int koff = hb * 8;
        #pragma unroll
        for (int kf = 0; kf < 4; ++kf) {
            v8h lo = *(const v8h*)&hlds[n16 * U_DIM + kf * 32 + koff];
            v8h hi = *(const v8h*)&hlds[n16 * U_DIM + kf * 32 + 16 + koff];
            #pragma unroll
            for (int i = 0; i < 8; ++i) { a[kf][i] = lo[i]; a[kf][8 + i] = hi[i]; }
        }

        #pragma unroll
        for (int j = 0; j < 8; ++j) {
            v8f az, ar, ah;
            #pragma unroll
            for (int e = 0; e < 8; ++e) { az[e] = 0.0f; ar[e] = 0.0f; ah[e] = 0.0f; }

            // Software-pipelined B fragments: loads for kf+1 in flight during kf's WMMAs.
            v16h b0 = *(const v16h*)&wl[FID(j, 0, 0)];
            v16h b1 = *(const v16h*)&wl[FID(j, 1, 0)];
            v16h b2 = *(const v16h*)&wl[FID(j, 2, 0)];
            #pragma unroll
            for (int kf = 0; kf < 4; ++kf) {
                v16h p0, p1, p2;
                if (kf < 3) {
                    p0 = *(const v16h*)&wl[FID(j, 0, kf + 1)];
                    p1 = *(const v16h*)&wl[FID(j, 1, kf + 1)];
                    p2 = *(const v16h*)&wl[FID(j, 2, kf + 1)];
                }
                az = __builtin_amdgcn_wmma_f32_16x16x32_f16(false, a[kf], false, b0, (short)0, az, false, false);
                ar = __builtin_amdgcn_wmma_f32_16x16x32_f16(false, a[kf], false, b1, (short)0, ar, false, false);
                ah = __builtin_amdgcn_wmma_f32_16x16x32_f16(false, a[kf], false, b2, (short)0, ah, false, false);
                if (kf < 3) { b0 = p0; b1 = p1; b2 = p2; }
            }

            // Per-column gate constants for this j-group (two b128 LDS loads)
            v4f c0 = *(const v4f*)&cbase[j * 16 * 8];
            v4f c1 = *(const v4f*)&cbase[j * 16 * 8 + 4];
            float kz = c0[0], kr = c0[1], kh = c0[2], sz = c0[3];
            float sr = c1[0], b0h = c1[1], b1h = c1[2];

            #pragma unroll
            for (int e = 0; e < 8; ++e) {
                int M = e + hb * 8;
                int hidx = M * U_DIM + j * 16 + n16;
                float hold = hfw[hidx];           // f32 carry from LDS
                float xs = xv[e];
                float z  = fsigmoid(xs * kz + sz + az[e]);
                float r  = fsigmoid(xs * kr + sr + ar[e]);
                float hh = ftanh(xs * kh + b0h + r * (ah[e] + b1h));
                float hn = z * hold + (1.0f - z) * hh;
                hfw[hidx]  = hn;
                hlds[hidx] = (_Float16)hn;
            }
        }
    };

    // ---- EDL head: out = h @ w(128x4) + b; lanes split (row, {c, c+2}) ----
    auto dng = [&](int g) {
        const float* w = ek + g * U_DIM * 4;
        const int row = n16;
        const int c0  = hb;  // lanes<16 -> cols {0,2}; lanes>=16 -> cols {1,3}
        float s0 = eb[g * 4 + c0];
        float s1 = eb[g * 4 + c0 + 2];
        #pragma unroll 4
        for (int k = 0; k < U_DIM; ++k) {
            float hv = hfw[row * U_DIM + k];
            s0 += hv * w[k * 4 + c0];
            s1 += hv * w[k * 4 + c0 + 2];
        }
        float r0, r1;
        if (c0 == 0) {
            r0 = s0;                       // mu
            xcur[row] = s0;                // autoregressive feedback
            r1 = fsoftplus(s1) + 1.0f;     // softplus(la) + 1
        } else {
            r0 = fsoftplus(s0);            // softplus(lv)
            r1 = fsoftplus(s1);            // softplus(lb)
        }
        long ob = (long)(rowbase + row) * (GAMMA * 4) + g * 4;
        out[ob + c0]     = r0;
        out[ob + c0 + 2] = r1;
    };

    // ---- Warm-up: 48 steps over x[:, :48] ----
    #pragma unroll 1
    for (int t = 0; t < WIN - 1; ++t) {
        if (lane < 16) xcur[lane] = x[(long)(rowbase + lane) * WIN + t];
        gru_step();
    }

    // ---- Autoregressive: head 0 on warm h, then 27 (step + head) ----
    dng(0);
    #pragma unroll 1
    for (int g = 1; g < GAMMA; ++g) {
        gru_step();
        dng(g);
    }
}

extern "C" void kernel_launch(void* const* d_in, const int* in_sizes, int n_in,
                              void* d_out, int out_size, void* d_ws, size_t ws_size,
                              hipStream_t stream) {
    const float* x   = (const float*)d_in[0];
    const float* gk  = (const float*)d_in[1];
    const float* grk = (const float*)d_in[2];
    const float* gb  = (const float*)d_in[3];
    const float* ek  = (const float*)d_in[4];
    const float* eb  = (const float*)d_in[5];
    float* out = (float*)d_out;

    size_t smem = (size_t)NFRAG * FRAG_HALF * 2              // swizzled W_rec (f16)       96 KB
                + (size_t)WAVES * ROWS_PER_WAVE * U_DIM * 2  // per-wave h (f16)           16 KB
                + (size_t)WAVES * ROWS_PER_WAVE * U_DIM * 4  // per-wave h (f32 carry)     32 KB
                + (size_t)U_DIM * 8 * 4                      // packed gate constants       4 KB
                + (size_t)WAVES * ROWS_PER_WAVE * 4;         // per-wave x feed            256 B

    hipFuncSetAttribute((const void*)fi_ernn_kernel,
                        hipFuncAttributeMaxDynamicSharedMemorySize, (int)smem);

    dim3 grid(BATCH / ROWS_PER_BLOCK);   // 256 workgroups
    dim3 block(WAVES * 32);              // 4 waves
    hipLaunchKernelGGL(fi_ernn_kernel, grid, block, smem, stream,
                       x, gk, grk, gb, ek, eb, out);
}
